// MTRNN_10505490006713
// MI455X (gfx1250) — matrix operand order repaired and checked by reading, not compiled, not run on hardware
//
#include <hip/hip_runtime.h>

// ---------------------------------------------------------------------------
// MTRNN (4-cell multi-timescale LSTM) for MI455X / gfx1250.
// Critical path = 896 sequential W_hh mat-vecs (8192x2048). Strategy:
//   * bf16 weights (halves per-step L2 traffic; 33.5MB/cell << 192MB L2)
//   * W_ih @ x hoisted off the critical path as batched bf16 WMMA GEMMs
//   * recurrent mat-vec also via v_wmma_f32_16x16x32_bf16 (h broadcast in A)
//   * fp32 accumulation + fp32 gating math
// ---------------------------------------------------------------------------

typedef __attribute__((ext_vector_type(16))) __bf16 v16bf;
typedef __attribute__((ext_vector_type(8)))  __bf16 v8bf;
typedef __attribute__((ext_vector_type(8)))  float  v8f;

#define HDIM 2048
#define GDIM 8192          // 4 * HDIM (i,f,g,o gate rows)
#define KSPLIT 8           // waves per step-block splitting the K dimension
#define KW (HDIM / KSPLIT) // 256

static __device__ __forceinline__ v16bf join8(v8bf lo, v8bf hi) {
  return __builtin_shufflevector(lo, hi, 0,1,2,3,4,5,6,7,8,9,10,11,12,13,14,15);
}

static __device__ __forceinline__ v8f wmma_bf16(v16bf a, v16bf b, v8f c) {
  return __builtin_amdgcn_wmma_f32_16x16x32_bf16(
      /*neg_a=*/false, a, /*neg_b=*/false, b,
      /*c_mod=*/(short)0, c, /*reuse_a=*/false, /*reuse_b=*/false);
}

// -------------------------- utility kernels --------------------------------

__global__ void cvt_f32_bf16(const float* __restrict__ s,
                             __bf16* __restrict__ d, size_t n) {
  size_t i = (size_t)blockIdx.x * blockDim.x + threadIdx.x;
  if (i < n) d[i] = (__bf16)s[i];
}

// strided row gather + convert: d[r, c] = (bf16) s[(roff + r*rstride), c]
__global__ void cvt_rows_f32_bf16(const float* __restrict__ s,
                                  __bf16* __restrict__ d,
                                  int rows, int cols, int roff, int rstride) {
  int i = blockIdx.x * blockDim.x + threadIdx.x;
  if (i < rows * cols) {
    int r = i / cols, c = i % cols;
    d[i] = (__bf16)s[(size_t)(roff + r * rstride) * cols + c];
  }
}

__global__ void bias_sum(const float* __restrict__ a,
                         const float* __restrict__ b,
                         float* __restrict__ o, int n) {
  int i = blockIdx.x * blockDim.x + threadIdx.x;
  if (i < n) o[i] = a[i] + b[i];
}

__global__ void init_state(float* __restrict__ c,
                           __bf16* __restrict__ h0,
                           __bf16* __restrict__ h1, int n) {
  int i = blockIdx.x * blockDim.x + threadIdx.x;
  if (i < n) { c[i] = 0.f; h0[i] = (__bf16)0.f; h1[i] = (__bf16)0.f; }
}

__global__ void copy_vec(const float* __restrict__ s, float* __restrict__ d, int n) {
  int i = blockIdx.x * blockDim.x + threadIdx.x;
  if (i < n) d[i] = s[i];
}

// -------------------- batched GEMM: C = A @ B^T (bf16 -> f32) --------------
// A: (M,K) row-major bf16, B: (N,K) row-major bf16, C: (M,N) f32.
// One wave computes a 16x64 output tile (A fragment reused across 4 B tiles).
// Fragments A 16x32, B 32x16 use contiguous k = 16*half + s per lane; D 16x16 per CDNA5 ISA 7.12.2.
__global__ __launch_bounds__(256)
void gemm_nt_bf16(const __bf16* __restrict__ A, const __bf16* __restrict__ B,
                  float* __restrict__ C, int M, int N, int K) {
  int lane = threadIdx.x & 31;
  int wid  = blockIdx.x * (blockDim.x >> 5) + (threadIdx.x >> 5);
  int tilesN = N >> 6;                 // 64-wide N tiles
  int tm = (wid / tilesN) << 4;
  int tn = (wid % tilesN) << 6;
  if (tm >= M) return;                 // whole-wave uniform exit (EXEC stays full)

  int mrow = lane & 15;
  int half = lane >> 4;

  v8f zero = {};
  v8f acc[4] = {zero, zero, zero, zero};

  const __bf16* ap = A + (size_t)(tm + mrow) * K + half * 16;  // A: 16 contiguous K per lane
  const __bf16* bp = B + (size_t)(tn + mrow) * K + half * 16;  // B: 16 contiguous K per lane

  for (int k0 = 0; k0 < K; k0 += 32) {
    v8bf alo = *(const v8bf*)(ap + k0);
    v8bf ahi = *(const v8bf*)(ap + k0 + 8);
    v16bf a = join8(alo, ahi);
#pragma unroll
    for (int s = 0; s < 4; ++s) {
      v16bf b = *(const v16bf*)(bp + (size_t)s * 16 * K + k0);
      acc[s] = wmma_bf16(a, b, acc[s]);
    }
  }
#pragma unroll
  for (int s = 0; s < 4; ++s)
#pragma unroll
    for (int r = 0; r < 8; ++r) {
      int row = tm + r + half * 8;
      int col = tn + s * 16 + (lane & 15);
      C[(size_t)row * N + col] = acc[s][r];
    }
}

// ------------------------- recurrent LSTM step -----------------------------
// Block = 256 thr (8 waves), covers 16 hidden units. Each wave takes a K-slice
// of 256 and accumulates 4 gate dot-product groups via WMMA (h broadcast into
// the A matrix; 16 W_hh rows as B columns; D row 0 = the 16 dot products).
// Partials reduced in LDS; wave0/lanes 0-15 do the fp32 gating.
__global__ __launch_bounds__(256)
void lstm_step(const float* __restrict__ Z, int t,
               const __bf16* __restrict__ Whh,
               const float* __restrict__ bsum,
               float* __restrict__ c,
               const __bf16* __restrict__ hprev,
               __bf16* __restrict__ hnext,
               float* __restrict__ hs_out) {
  __shared__ float red[KSPLIT * 4 * 16];

  int lane = threadIdx.x & 31;
  int wave = threadIdx.x >> 5;
  int j0   = blockIdx.x << 4;      // 16 hidden units per block (grid = HDIM/16)
  int n    = lane & 15;
  int half = lane >> 4;

  v8f zero = {};
  v8f acc[4] = {zero, zero, zero, zero};

  int kbase = wave * KW;
  for (int kc = 0; kc < KW; kc += 32) {
    int k0 = kbase + kc;
    // A fragment: h broadcast into every M row (only D row 0 is consumed)
    const __bf16* hp = hprev + k0 + half * 16;
    v16bf a = join8(*(const v8bf*)hp, *(const v8bf*)(hp + 8));
#pragma unroll
    for (int g = 0; g < 4; ++g) {
      const __bf16* wp = Whh + (size_t)(g * HDIM + j0 + n) * HDIM + k0 + half * 16;
      v16bf b = *(const v16bf*)wp;
      acc[g] = wmma_bf16(a, b, acc[g]);
    }
  }

  // D[0, n] lives in VGPR0 of lanes 0-15 (lane n -> hidden unit j0+n)
  if (lane < 16) {
#pragma unroll
    for (int g = 0; g < 4; ++g)
      red[(wave * 4 + g) * 16 + n] = acc[g][0];
  }
  __syncthreads();

  if (threadIdx.x < 16) {
    int j = j0 + n;
    float m[4];
#pragma unroll
    for (int g = 0; g < 4; ++g) {
      float s = 0.f;
#pragma unroll
      for (int w = 0; w < KSPLIT; ++w) s += red[(w * 4 + g) * 16 + n];
      m[g] = s + Z[(size_t)t * GDIM + g * HDIM + j] + bsum[g * HDIM + j];
    }
    // torch gate order: i, f, g, o
    float ig = 1.f / (1.f + __expf(-m[0]));
    float fg = 1.f / (1.f + __expf(-m[1]));
    float gg = tanhf(m[2]);
    float og = 1.f / (1.f + __expf(-m[3]));
    float cn = fg * c[j] + ig * gg;
    float hn = og * tanhf(cn);
    hn = fmaxf(hn, 0.f);   // extra ReLU on h
    cn = fmaxf(cn, 0.f);   // extra ReLU on c
    c[j] = cn;
    hnext[j] = (__bf16)hn;
    hs_out[(size_t)t * HDIM + j] = hn;
  }
}

// ------------------------------ host driver --------------------------------

extern "C" void kernel_launch(void* const* d_in, const int* in_sizes, int n_in,
                              void* d_out, int out_size, void* d_ws, size_t ws_size,
                              hipStream_t stream) {
  (void)in_sizes; (void)n_in; (void)out_size; (void)ws_size;

  const float* x = (const float*)d_in[0];

  char* ws = (char*)d_ws;
  size_t off = 0;
  auto alloc = [&](size_t bytes) -> char* {
    char* p = ws + off;
    off += (bytes + 255) & ~(size_t)255;
    return p;
  };

  const size_t WN = (size_t)GDIM * HDIM;           // 16.7M weight elements
  __bf16* wI  = (__bf16*)alloc(WN * 2);            // W_ih (bf16, reused per cell)
  __bf16* wH  = (__bf16*)alloc(WN * 2);            // W_hh (bf16, reused per cell)
  __bf16* xb  = (__bf16*)alloc((size_t)512 * HDIM * 2);
  float*  Zb  = (float*) alloc((size_t)512 * GDIM * 4);
  float*  hs1 = (float*) alloc((size_t)512 * HDIM * 4);
  float*  hs2 = (float*) alloc((size_t)128 * HDIM * 4);
  float*  hs3 = (float*) alloc((size_t)128 * HDIM * 4);
  float*  hs4 = (float*) alloc((size_t)128 * HDIM * 4);
  float*  bs  = (float*) alloc((size_t)GDIM * 4);
  float*  cst = (float*) alloc((size_t)HDIM * 4);
  __bf16* hp0 = (__bf16*)alloc((size_t)HDIM * 2);
  __bf16* hp1 = (__bf16*)alloc((size_t)HDIM * 2);

  for (int cell = 0; cell < 4; ++cell) {
    const float* Wi = (const float*)d_in[1 + 4 * cell + 0];
    const float* Wh = (const float*)d_in[1 + 4 * cell + 1];
    const float* bi = (const float*)d_in[1 + 4 * cell + 2];
    const float* bh = (const float*)d_in[1 + 4 * cell + 3];
    const int T = (cell == 0) ? 512 : 128;

    // Weights -> bf16 (L2-resident working set for the recurrence)
    cvt_f32_bf16<<<(unsigned)((WN + 255) / 256), 256, 0, stream>>>(Wi, wI, WN);
    cvt_f32_bf16<<<(unsigned)((WN + 255) / 256), 256, 0, stream>>>(Wh, wH, WN);
    bias_sum<<<GDIM / 256, 256, 0, stream>>>(bi, bh, bs, GDIM);

    // Phase input -> bf16 (cell2 consumes hs1[1::4])
    if (cell == 0)
      cvt_rows_f32_bf16<<<(512 * HDIM) / 256, 256, 0, stream>>>(x,   xb, 512, HDIM, 0, 1);
    else if (cell == 1)
      cvt_rows_f32_bf16<<<(128 * HDIM) / 256, 256, 0, stream>>>(hs1, xb, 128, HDIM, 1, 4);
    else if (cell == 2)
      cvt_rows_f32_bf16<<<(128 * HDIM) / 256, 256, 0, stream>>>(hs2, xb, 128, HDIM, 0, 1);
    else
      cvt_rows_f32_bf16<<<(128 * HDIM) / 256, 256, 0, stream>>>(hs3, xb, 128, HDIM, 0, 1);

    init_state<<<HDIM / 256, 256, 0, stream>>>(cst, hp0, hp1, HDIM);

    // Off-critical-path batched GEMM: Z = X @ W_ih^T  (T x 8192)
    int waves = (T / 16) * (GDIM / 64);
    gemm_nt_bf16<<<waves / 8, 256, 0, stream>>>(xb, wI, Zb, T, GDIM, HDIM);

    // Sequential recurrence (kernel boundary = grid-wide sync; ping-pong h)
    float* hsOut = (cell == 0) ? hs1 : (cell == 1) ? hs2 : (cell == 2) ? hs3 : hs4;
    for (int t = 0; t < T; ++t) {
      const __bf16* hp = (t & 1) ? hp1 : hp0;
      __bf16*       hn = (t & 1) ? hp0 : hp1;
      lstm_step<<<HDIM / 16, 256, 0, stream>>>(Zb, t, wH, bs, cst, hp, hn, hsOut);
    }
  }

  copy_vec<<<HDIM / 256, 256, 0, stream>>>(hs4 + (size_t)127 * HDIM,
                                           (float*)d_out, HDIM);
}
